// HPUAttentionImpl_49289044689579
// MI455X (gfx1250) — matrix operand order repaired and checked by reading, not compiled, not run on hardware
//
#include <hip/hip_runtime.h>
#include <hip/hip_bf16.h>
#include <stdint.h>

// Problem constants (match reference)
#define B_N       2
#define S_N       1024
#define H_N       32
#define KV_N      8
#define D_N       128
#define G_N       (H_N / KV_N)
#define BLK_N     128
#define NBLOCKS_N 32
#define SCALE_F   0.08838834764831845f   // 1/sqrt(128)

typedef __attribute__((ext_vector_type(16))) _Float16 v16h;
typedef __attribute__((ext_vector_type(8)))  float    v8f;

// ---------------------------------------------------------------------------
// KV-cache: copy base cache, then scatter prompt blocks by block_indices
// ---------------------------------------------------------------------------
__global__ void cache_copy(const float4* __restrict__ kvc,
                           float4* __restrict__ kc,
                           float4* __restrict__ vc, int n4)
{
    int i = blockIdx.x * blockDim.x + threadIdx.x;
    if (i < n4) {
        kc[i] = kvc[i];          // kv_cache[0] -> key_cache
        vc[i] = kvc[n4 + i];     // kv_cache[1] -> value_cache
    }
}

__global__ void cache_scatter(const float4* __restrict__ key4,
                              const float4* __restrict__ val4,
                              const int* __restrict__ bidx,
                              float4* __restrict__ kc,
                              float4* __restrict__ vc, int be4)
{
    int e = blockIdx.x * blockDim.x + threadIdx.x;   // element within block
    int blk = blockIdx.y;                            // source prompt block
    if (e < be4) {
        int dst = bidx[blk];
        kc[(size_t)dst * be4 + e] = key4[(size_t)blk * be4 + e];
        vc[(size_t)dst * be4 + e] = val4[(size_t)blk * be4 + e];
    }
}

// ---------------------------------------------------------------------------
// Flash attention: one wave32 per (batch, head, 16-query tile)
// ---------------------------------------------------------------------------
__global__ __launch_bounds__(32)
void attn_fwd(const float* __restrict__ Q, const float* __restrict__ K,
              const float* __restrict__ V, const float* __restrict__ BIAS,
              const float* __restrict__ SLOPES, float* __restrict__ OUT)
{
    __shared__ __align__(16) _Float16 pbuf[16 * 32];   // P transpose staging

    const int lane = threadIdx.x;
    const int n    = lane & 15;         // column / row-within-16 id
    const int hi   = lane >> 4;         // lane half
    const int q0   = blockIdx.x * 16;   // query tile start
    const int h    = blockIdx.y;        // query head
    const int b    = blockIdx.z;        // batch
    const int kvh  = h / G_N;           // GQA kv head
    const float slope = SLOPES[h];

    // ---- Q tile (16x128) as 4 A-fragments (16x32 f16), pre-scaled ----
    // A layout: lane row m = lane&15; K ids: lo-half {0-7,16-23}, hi-half {8-15,24-31}
    v16h aq[4];
    {
        const float* qrow = Q + ((size_t)b * S_N + (q0 + n)) * (H_N * D_N)
                              + (size_t)h * D_N;
        #pragma unroll
        for (int c = 0; c < 4; ++c) {
            const int d0 = c * 32 + hi * 8;
            #pragma unroll
            for (int t = 0; t < 8; ++t) {
                aq[c][t]     = (_Float16)(qrow[d0 + t]      * SCALE_F);
                aq[c][t + 8] = (_Float16)(qrow[d0 + 16 + t] * SCALE_F);
            }
        }
    }

    // Flash state (C layout: element i -> row i + 8*hi, col n)
    float m_i[8], l_i[8];
    v8f   o_acc[8];                     // 16x128 output tile, 8 col-chunks of 16
    #pragma unroll
    for (int i = 0; i < 8; ++i) { m_i[i] = -__builtin_inff(); l_i[i] = 0.0f; }
    #pragma unroll
    for (int dc = 0; dc < 8; ++dc)
        #pragma unroll
        for (int e = 0; e < 8; ++e) o_acc[dc][e] = 0.0f;

    const int jmax = ((q0 + 15) >> 5) << 5;          // last 32-key tile (causal)
    for (int j0 = 0; j0 <= jmax; j0 += 32) {

        // ---- prefetch next tile's K/V lines (global_prefetch_b8) ----
        // 32 rows/tile, 512B used per row = 4 x 128B lines; one prefetch
        // instruction covers 32 lanes x 1 line. Uniform guard -> no divergence.
        if (j0 + 32 <= jmax) {
            const int jn = j0 + 32;
            const char* pk = (const char*)(K + ((size_t)b * S_N + jn) * (KV_N * D_N)
                                             + (size_t)kvh * D_N);
            const char* pv = (const char*)(V + ((size_t)b * S_N + jn) * (KV_N * D_N)
                                             + (size_t)kvh * D_N);
            const size_t lo = (size_t)lane * (KV_N * D_N * 4);   // row stride 4KB
            #pragma unroll
            for (int l = 0; l < 4; ++l) {
                __builtin_prefetch(pk + lo + l * 128, 0, 3);
                __builtin_prefetch(pv + lo + l * 128, 0, 3);
            }
        }

        // ---- scores: two 16x16 tiles, contraction over D=128 (4 WMMAs each) ----
        v8f s[2];
        #pragma unroll
        for (int jt = 0; jt < 2; ++jt) {
            v8f acc = {};
            const int key = j0 + jt * 16 + n;        // B-frag column = key
            const float* krow = K + ((size_t)b * S_N + key) * (KV_N * D_N)
                                  + (size_t)kvh * D_N;
            #pragma unroll
            for (int c = 0; c < 4; ++c) {
                v16h bk;                              // B layout: k = 8*hi + t (t<8), 16 + 8*hi + (t-8) (t>=8)
                const int d0 = c * 32 + hi * 8;
                #pragma unroll
                for (int t = 0; t < 16; ++t) bk[t] = (_Float16)krow[d0 + t + (t >= 8 ? 8 : 0)];
                acc = __builtin_amdgcn_wmma_f32_16x16x32_f16(
                          false, aq[c], false, bk, (short)0, acc, false, false);
            }
            // bias: causal mask from attn_bias + alibi slope*(k - q)
            #pragma unroll
            for (int i = 0; i < 8; ++i) {
                const int qr = q0 + i + 8 * hi;
                const int kp = j0 + jt * 16 + n;
                acc[i] += BIAS[((size_t)b * S_N + qr) * S_N + kp]
                        + slope * (float)(kp - qr);
            }
            s[jt] = acc;
        }

        // ---- online softmax update (row stats across 16-lane halves) ----
        float alpha[8];
        #pragma unroll
        for (int i = 0; i < 8; ++i) {
            float tmax = fmaxf(s[0][i], s[1][i]);
            #pragma unroll
            for (int msk = 1; msk < 16; msk <<= 1)
                tmax = fmaxf(tmax, __shfl_xor(tmax, msk, 32));
            const float mnew = fmaxf(m_i[i], tmax);
            alpha[i] = __expf(m_i[i] - mnew);
            m_i[i] = mnew;
            const float p0 = __expf(s[0][i] - mnew);
            const float p1 = __expf(s[1][i] - mnew);
            s[0][i] = p0; s[1][i] = p1;
            float rs = p0 + p1;
            #pragma unroll
            for (int msk = 1; msk < 16; msk <<= 1)
                rs += __shfl_xor(rs, msk, 32);
            l_i[i] = l_i[i] * alpha[i] + rs;
        }
        #pragma unroll
        for (int dc = 0; dc < 8; ++dc)
            #pragma unroll
            for (int i = 0; i < 8; ++i) o_acc[dc][i] *= alpha[i];

        // ---- P: C layout -> LDS -> A-fragment layout (16x32 f16) ----
        __syncthreads();   // WAR fence vs previous iteration's reads (S_NOP, 1 wave)
        #pragma unroll
        for (int jt = 0; jt < 2; ++jt)
            #pragma unroll
            for (int i = 0; i < 8; ++i)
                pbuf[(i + 8 * hi) * 32 + jt * 16 + n] = (_Float16)s[jt][i];
        __syncthreads();
        union { v16h v; uint4 u[2]; } ap;
        ap.u[0] = *(const uint4*)&pbuf[n * 32 + hi * 8];        // K = koff+0..7
        ap.u[1] = *(const uint4*)&pbuf[n * 32 + hi * 8 + 16];   // K = koff+16..23

        // ---- O += P @ V : 8 WMMAs over the 8 d-chunks ----
        const float* vcol = V + ((size_t)b * S_N + (j0 + 8 * hi)) * (KV_N * D_N)
                              + (size_t)kvh * D_N + n;
        #pragma unroll
        for (int dc = 0; dc < 8; ++dc) {
            v16h bv;                                 // B[k][n] = V[j0+k][dc*16+n]
            #pragma unroll
            for (int t = 0; t < 16; ++t)
                bv[t] = (_Float16)vcol[(size_t)(t + (t >= 8 ? 8 : 0)) * (KV_N * D_N) + dc * 16];
            o_acc[dc] = __builtin_amdgcn_wmma_f32_16x16x32_f16(
                            false, ap.v, false, bv, (short)0, o_acc[dc], false, false);
        }
    }

    // ---- epilogue: divide by running sum, store ----
    #pragma unroll
    for (int i = 0; i < 8; ++i) {
        const float inv = 1.0f / l_i[i];
        const int qr = q0 + i + 8 * hi;
        float* orow = OUT + ((size_t)b * S_N + qr) * (H_N * D_N) + (size_t)h * D_N;
        #pragma unroll
        for (int dc = 0; dc < 8; ++dc)
            orow[dc * 16 + n] = o_acc[dc][i] * inv;
    }
}

// ---------------------------------------------------------------------------
extern "C" void kernel_launch(void* const* d_in, const int* in_sizes, int n_in,
                              void* d_out, int out_size, void* d_ws, size_t ws_size,
                              hipStream_t stream)
{
    (void)n_in; (void)out_size; (void)d_ws; (void)ws_size;

    const float* query  = (const float*)d_in[0];
    const float* key    = (const float*)d_in[1];
    const float* value  = (const float*)d_in[2];
    const float* kvc    = (const float*)d_in[3];
    const int*   bidx   = (const int*)  d_in[4];
    const float* bias   = (const float*)d_in[5];
    const float* slopes = (const float*)d_in[6];

    float* out = (float*)d_out;
    float* key_cache = out + (size_t)B_N * S_N * H_N * D_N;
    float* val_cache = key_cache + (size_t)NBLOCKS_N * BLK_N * KV_N * D_N;

    // 1) base cache copy (vectorized float4)
    const int n4 = NBLOCKS_N * BLK_N * KV_N * D_N / 4;
    cache_copy<<<(n4 + 255) / 256, 256, 0, stream>>>(
        (const float4*)kvc, (float4*)key_cache, (float4*)val_cache, n4);

    // 2) scatter prompt blocks into cache
    const int nblk = in_sizes[4];                    // B*S/BLK
    const int be4  = BLK_N * KV_N * D_N / 4;
    dim3 sgrid((be4 + 255) / 256, nblk);
    cache_scatter<<<sgrid, 256, 0, stream>>>(
        (const float4*)key, (const float4*)value, bidx,
        (float4*)key_cache, (float4*)val_cache, be4);

    // 3) attention
    dim3 agrid(S_N / 16, H_N, B_N);
    attn_fwd<<<agrid, 32, 0, stream>>>(query, key, value, bias, slopes, out);
}
